// RelativeMultiHeadAttention_15049565405876
// MI455X (gfx1250) — matrix operand order repaired and checked by reading, not compiled, not run on hardware
//
#include <hip/hip_runtime.h>

// ---------------------------------------------------------------------------
// RelativeMultiHeadAttention (Transformer-XL style) for gfx1250 / MI455X.
// All matmuls run on v_wmma_f32_16x16x32_bf16 (wave32 WMMA), f32 accumulate.
// ---------------------------------------------------------------------------

typedef __bf16 bf16_t;
typedef __bf16 bfx4  __attribute__((ext_vector_type(4)));
typedef __bf16 bfx8  __attribute__((ext_vector_type(8)));
typedef __bf16 v16bf __attribute__((ext_vector_type(16)));
typedef float  v8f   __attribute__((ext_vector_type(8)));

#define DMODEL 512
#define NHEADS 16
#define DHEAD  32
#define SEQ    1024
#define BATCH  4
#define MTOT   (BATCH * SEQ)   // 4096 rows

// A-matrix 16x32 bf16 fragment: lane L holds M=L%16; K = 16*(L/16)..+15 contiguous.
__device__ __forceinline__ v16bf frag_a(const bf16_t* p0, const bf16_t* p1) {
  (void)p1;
  return *(const v16bf*)p0;
}
// B-matrix 32x16 bf16 fragment: lane holds column n=lane%16, K = 16*(lane/16)..+15
// contiguous -> single 32B read.
__device__ __forceinline__ v16bf frag_b(const bf16_t* p) {
  return *(const v16bf*)p;
}
__device__ __forceinline__ v16bf zero_frag() {
  v16bf z;
#pragma unroll
  for (int i = 0; i < 16; ++i) z[i] = (bf16_t)0.0f;
  return z;
}
__device__ __forceinline__ v8f wmma_bf16(v16bf a, v16bf b, v8f c) {
  return __builtin_amdgcn_wmma_f32_16x16x32_bf16(false, a, false, b, (short)0, c,
                                                 false, false);
}

// ---------------------------------------------------------------------------
// Generic projection GEMM: Y[4096,512] = A[4096,512] x W[512,512] (+ bias)
// block = 256 threads (8 waves); WG tile 128(M) x 64(N); K-step 32.
// Wave w owns M-strip w*16..w*16+15 and all four 16-wide N subtiles:
// one A fragment is reused across 4 WMMAs per K-step.
// mode 0: q proj   -> out0 = bf16(Y + u_bias), out1 = bf16(Y + v_bias)
// mode 1: k proj   -> out0 = bf16(Y)
// mode 2: v proj   -> out0 = bf16(Y) stored TRANSPOSED as [B,H,Dh,T]
// mode 3: pos proj -> out0 = bf16(Y)           (no bias)
// mode 4: final    -> outf = Y (fp32, + bias)
// ---------------------------------------------------------------------------
__global__ void proj_gemm_kernel(const float* __restrict__ A,
                                 const float* __restrict__ W,
                                 const float* __restrict__ bias,
                                 const float* __restrict__ ubias,
                                 const float* __restrict__ vbias,
                                 bf16_t* __restrict__ out0,
                                 bf16_t* __restrict__ out1,
                                 float*  __restrict__ outf,
                                 int mode) {
  __shared__ bf16_t ldsA[128][32];  // A tile, row-major        (8 KB)
  __shared__ bf16_t ldsB[64][32];   // W tile, stored [n][k]    (4 KB)

  const int tid  = threadIdx.x;
  const int wave = tid >> 5;
  const int lane = tid & 31;
  const int m0 = blockIdx.y * 128;
  const int n0 = blockIdx.x * 64;
  const int mL   = lane & 15;
  const int kh8  = (lane >> 4) * 16;   // A-frag K base
  const int kb16 = (lane >> 4) * 16;   // B-frag K base

  v8f acc[4] = {{}, {}, {}, {}};
  for (int kk = 0; kk < DMODEL; kk += 32) {
    // ---- stage A tile 128x32 (fp32 -> bf16) ----
    const int am   = tid >> 1;
    const int hsel = (tid & 1) * 16;
    {
      const float* src = A + (size_t)(m0 + am) * DMODEL + kk + hsel;
#pragma unroll
      for (int i = 0; i < 4; ++i) {
        float4 f = *(const float4*)(src + i * 4);
        bfx4 w;
        w[0] = (bf16_t)f.x; w[1] = (bf16_t)f.y;
        w[2] = (bf16_t)f.z; w[3] = (bf16_t)f.w;
        *(bfx4*)&ldsA[am][hsel + i * 4] = w;
      }
    }
    // ---- stage W tile 32x64 transposed -> ldsB[n][k] ----
    const int bk = tid >> 3;
    const int n8 = (tid & 7) * 8;
    {
      const float* srcW = W + (size_t)(kk + bk) * DMODEL + n0 + n8;
#pragma unroll
      for (int i = 0; i < 2; ++i) {
        float4 f = *(const float4*)(srcW + i * 4);
        ldsB[n8 + i * 4 + 0][bk] = (bf16_t)f.x;
        ldsB[n8 + i * 4 + 1][bk] = (bf16_t)f.y;
        ldsB[n8 + i * 4 + 2][bk] = (bf16_t)f.z;
        ldsB[n8 + i * 4 + 3][bk] = (bf16_t)f.w;
      }
    }
    // prefetch next K-step's global lines into cache (global_prefetch_b8)
    if (kk + 32 < DMODEL) {
      __builtin_prefetch(A + (size_t)(m0 + am) * DMODEL + kk + 32 + hsel, 0, 1);
      __builtin_prefetch(W + (size_t)(kk + 32 + bk) * DMODEL + n0 + n8, 0, 1);
    }
    __syncthreads();
    v16bf a = frag_a(&ldsA[wave * 16 + mL][kh8], &ldsA[wave * 16 + mL][kh8 + 16]);
#pragma unroll
    for (int g = 0; g < 4; ++g) {
      v16bf b = frag_b(&ldsB[g * 16 + mL][kb16]);
      acc[g] = wmma_bf16(a, b, acc[g]);
    }
    __syncthreads();
  }

  // ---- epilogue ----
#pragma unroll
  for (int g = 0; g < 4; ++g) {
    const int gn = n0 + g * 16 + mL;
    float bval = bias ? bias[gn] : 0.0f;
    float uv = 0.0f, vv = 0.0f;
    if (mode == 0) { uv = ubias[gn]; vv = vbias[gn]; }  // [H,Dh] flat == col idx
#pragma unroll
    for (int r = 0; r < 8; ++r) {
      int gm = m0 + wave * 16 + r + 8 * (lane >> 4);
      float y = acc[g][r] + bval;
      if (mode == 0) {
        out0[(size_t)gm * DMODEL + gn] = (bf16_t)(y + uv);
        out1[(size_t)gm * DMODEL + gn] = (bf16_t)(y + vv);
      } else if (mode == 1 || mode == 3) {
        out0[(size_t)gm * DMODEL + gn] = (bf16_t)y;
      } else if (mode == 2) {
        int bb = gm >> 10, t = gm & (SEQ - 1);
        int hh = gn >> 5,  d = gn & (DHEAD - 1);
        out0[(size_t)((bb * NHEADS + hh) * DHEAD + d) * SEQ + t] = (bf16_t)y;
      } else {
        outf[(size_t)gm * DMODEL + gn] = y;
      }
    }
  }
}

// ---------------------------------------------------------------------------
// Fused attention: one 4-wave WG per (b, h, 16-query-row block).
// score = (QuK^T + relshift(QvP^T)) / sqrt(D); softmax; ctx = attn @ V.
// relshift(P)[t,s] = P[t, s-t+T-1]   (s <= t)
//                  = 0               (s == t+1)
//                  = P[t+1, s-t-2]   (s >  t+1)
// Waves split the key-tile loop (wave-private P spill, no block barrier in the
// hot loop), split softmax rows, and split the attn@V K dimension with an LDS
// reduction.
// ---------------------------------------------------------------------------
__global__ void attn_kernel(const bf16_t* __restrict__ qhu,
                            const bf16_t* __restrict__ qhv,
                            const bf16_t* __restrict__ khm,
                            const bf16_t* __restrict__ phm,
                            const bf16_t* __restrict__ vhT,  // [B,H,Dh,T]
                            float* __restrict__ ctx) {
  extern __shared__ char smem[];
  float*  S     = (float*)smem;                                 // [16][1024]
  bf16_t* Pr    = (bf16_t*)(smem + 16 * SEQ * 4);               // [16][1024]
  float*  pbuf  = (float*)(smem + 16 * SEQ * 4 + 16 * SEQ * 2); // [4][16][64]
  float*  red   = pbuf;                                         // [4][16][32] (reuse)
  float*  recip = pbuf + 4 * 16 * 64;                           // [16]

  const int tid  = threadIdx.x;
  const int wave = tid >> 5;
  const int lane = tid & 31;
  const int b  = blockIdx.z, h = blockIdx.y;
  const int t0 = blockIdx.x * 16;
  const int mL    = lane & 15;
  const int kh8   = (lane >> 4) * 16;
  const int kb16  = (lane >> 4) * 16;
  const int rbase = 8 * (lane >> 4);
  const float scale = 0.04419417382415922f;  // 1/sqrt(512)
  float* mypb = pbuf + wave * (16 * 64);

  // A-fragments (row = query index, K = head dim, Dh==32 -> exactly one WMMA)
  const size_t rowQ = ((size_t)(b * SEQ + t0 + mL)) * DMODEL + h * DHEAD;
  v16bf aU  = frag_a(qhu + rowQ + kh8, qhu + rowQ + kh8 + 16);
  v16bf aV0 = frag_a(qhv + rowQ + kh8, qhv + rowQ + kh8 + 16);
  int  r1  = t0 + mL + 1;
  bool ok1 = (r1 < SEQ);
  const size_t rowQ1 = ((size_t)(b * SEQ + (ok1 ? r1 : 0))) * DMODEL + h * DHEAD;
  v16bf aV1 = frag_a(qhv + rowQ1 + kh8, qhv + rowQ1 + kh8 + 16);
  if (!ok1) aV1 = zero_frag();

  // ---- phase 1: scores. Wave w handles key tiles s0 = (iter*4 + w)*16 ----
  for (int it = 0; it < SEQ / (16 * 4); ++it) {
    const int s0 = (it * 4 + wave) * 16;
    // content tile: Qu(16x32) x K^T(32x16); B-frag lane n = key row s0+n
    const size_t rowK = ((size_t)(b * SEQ + s0 + mL)) * DMODEL + h * DHEAD;
    v16bf bK = frag_b(khm + rowK + kb16);
    v8f c = {};
    c = wmma_bf16(aU, bK, c);

    // pos tiles: P cols [s0-t0+1008, +31] (rows t) and [s0-t0-17, +31] (rows t+1)
    v8f pa[2], pb2[2];
#pragma unroll
    for (int g = 0; g < 2; ++g) {
      int  cA  = s0 - t0 + (SEQ - 16) + g * 16 + mL;
      bool okA = (cA >= 0) && (cA < SEQ);
      v16bf bA = frag_b(phm + ((size_t)(b * SEQ + (okA ? cA : 0))) * DMODEL +
                        h * DHEAD + kb16);
      if (!okA) bA = zero_frag();
      v8f za = {};
      pa[g] = wmma_bf16(aV0, bA, za);

      int  cB  = s0 - t0 - 17 + g * 16 + mL;
      bool okB = (cB >= 0) && (cB < SEQ);
      v16bf bB = frag_b(phm + ((size_t)(b * SEQ + (okB ? cB : 0))) * DMODEL +
                        h * DHEAD + kb16);
      if (!okB) bB = zero_frag();
      v8f zb = {};
      pb2[g] = wmma_bf16(aV1, bB, zb);
    }
    // spill P tiles into the wave-private region for the anti-diagonal gather
#pragma unroll
    for (int r = 0; r < 8; ++r) {
      int mm = r + rbase;
#pragma unroll
      for (int g = 0; g < 2; ++g) {
        mypb[mm * 64 + g * 16 + mL]      = pa[g][r];
        mypb[mm * 64 + 32 + g * 16 + mL] = pb2[g][r];
      }
    }
    // wave-local: LDS is in-order per wave; fence keeps the compiler honest
    __threadfence_block();
    // combine content + shifted pos -> S
#pragma unroll
    for (int r = 0; r < 8; ++r) {
      int mm  = r + rbase;
      int s   = s0 + mL;
      int rel = s - (t0 + mm);
      int idx = mL - mm + 15;  // in [0,30]
      float pos = 0.0f;
      if (rel <= 0)     pos = mypb[mm * 64 + idx];
      else if (rel > 1) pos = mypb[mm * 64 + 32 + idx];
      S[mm * SEQ + s] = (c[r] + pos) * scale;
    }
    __threadfence_block();
  }
  __syncthreads();

  // ---- phase 2: row softmax, 4 rows per wave (wave32 shfl reductions) ----
  for (int j = 0; j < 4; ++j) {
    const int m = wave * 4 + j;
    float mx = -3.0e38f;
    for (int i = lane; i < SEQ; i += 32) mx = fmaxf(mx, S[m * SEQ + i]);
#pragma unroll
    for (int off = 16; off > 0; off >>= 1) mx = fmaxf(mx, __shfl_xor(mx, off, 32));
    float sum = 0.0f;
    for (int i = lane; i < SEQ; i += 32) {
      float e = __expf(S[m * SEQ + i] - mx);
      Pr[m * SEQ + i] = (bf16_t)e;
      sum += e;
    }
#pragma unroll
    for (int off = 16; off > 0; off >>= 1) sum += __shfl_xor(sum, off, 32);
    if (lane == 0) recip[m] = 1.0f / sum;
  }
  __syncthreads();

  // ---- phase 3: ctx(16x32) = attn(16x1024) @ V(1024x32), K split by wave ----
  v8f c0 = {}, c1 = {};
  const bf16_t* arow  = Pr + mL * SEQ;
  const size_t  vbase = ((size_t)(b * NHEADS + h)) * DHEAD * SEQ;
  const bf16_t* v0 = vhT + vbase + (size_t)(mL) * SEQ;        // d = n
  const bf16_t* v1 = vhT + vbase + (size_t)(16 + mL) * SEQ;   // d = 16 + n
  const int kbeg = wave * (SEQ / 4);
  for (int k0 = kbeg; k0 < kbeg + SEQ / 4; k0 += 32) {
    v16bf aP = frag_a(arow + k0 + kh8, arow + k0 + kh8 + 16);
    v16bf b0 = frag_b(v0 + k0 + kb16);
    v16bf b1 = frag_b(v1 + k0 + kb16);
    c0 = wmma_bf16(aP, b0, c0);
    c1 = wmma_bf16(aP, b1, c1);
  }
  // partials -> LDS: red[wave][m][d], d in [0,32)
#pragma unroll
  for (int r = 0; r < 8; ++r) {
    int mm = r + rbase;
    red[wave * 512 + mm * 32 + mL]      = c0[r];
    red[wave * 512 + mm * 32 + 16 + mL] = c1[r];
  }
  __syncthreads();
  // tree-reduce 4 partials, scale by 1/sum, store ctx. 128 thr x 4 elems.
  {
    const int m  = tid >> 3;
    const int d4 = (tid & 7) * 4;
    const float rc = recip[m];
    const size_t o = ((size_t)(b * SEQ + t0 + m)) * DMODEL + h * DHEAD;
#pragma unroll
    for (int jj = 0; jj < 4; ++jj) {
      int d = d4 + jj;
      float s = red[0 * 512 + m * 32 + d] + red[1 * 512 + m * 32 + d] +
                red[2 * 512 + m * 32 + d] + red[3 * 512 + m * 32 + d];
      ctx[o + d] = s * rc;
    }
  }
}

// ---------------------------------------------------------------------------
extern "C" void kernel_launch(void* const* d_in, const int* in_sizes, int n_in,
                              void* d_out, int out_size, void* d_ws, size_t ws_size,
                              hipStream_t stream) {
  (void)in_sizes; (void)n_in; (void)out_size; (void)ws_size;
  const float* q  = (const float*)d_in[0];
  const float* k  = (const float*)d_in[1];
  const float* v  = (const float*)d_in[2];
  const float* pe = (const float*)d_in[3];
  const float* Wq = (const float*)d_in[4];
  const float* bq = (const float*)d_in[5];
  const float* Wp = (const float*)d_in[6];
  const float* Wf = (const float*)d_in[7];
  const float* bf = (const float*)d_in[8];
  const float* ub = (const float*)d_in[9];
  const float* vb = (const float*)d_in[10];
  float* out = (float*)d_out;

  char* ws = (char*)d_ws;
  const size_t projBytes = (size_t)MTOT * DMODEL * sizeof(bf16_t);  // 4 MB each
  bf16_t* qhu = (bf16_t*)(ws);
  bf16_t* qhv = (bf16_t*)(ws + 1 * projBytes);
  bf16_t* khb = (bf16_t*)(ws + 2 * projBytes);
  bf16_t* phb = (bf16_t*)(ws + 3 * projBytes);
  bf16_t* vhT = (bf16_t*)(ws + 4 * projBytes);
  float*  ctx = (float*)(ws + 5 * projBytes);  // 8 MB fp32

  dim3 ggrid(DMODEL / 64, MTOT / 128);  // (8, 32)
  dim3 gblk(256);
  proj_gemm_kernel<<<ggrid, gblk, 0, stream>>>(q,  Wq, bq,      ub,      vb,
                                               qhu, qhv, nullptr, 0);
  proj_gemm_kernel<<<ggrid, gblk, 0, stream>>>(k,  Wq, bq,      nullptr, nullptr,
                                               khb, nullptr, nullptr, 1);
  proj_gemm_kernel<<<ggrid, gblk, 0, stream>>>(v,  Wq, bq,      nullptr, nullptr,
                                               vhT, nullptr, nullptr, 2);
  proj_gemm_kernel<<<ggrid, gblk, 0, stream>>>(pe, Wp, nullptr, nullptr, nullptr,
                                               phb, nullptr, nullptr, 3);

  size_t smem = (size_t)16 * SEQ * 4 + 16 * SEQ * 2 + 4 * 16 * 64 * 4 + 16 * 4;
  attn_kernel<<<dim3(SEQ / 16, NHEADS, BATCH), dim3(128), smem, stream>>>(
      qhu, qhv, khb, phb, vhT, ctx);

  proj_gemm_kernel<<<ggrid, gblk, 0, stream>>>(ctx, Wf, bf, nullptr, nullptr,
                                               nullptr, nullptr, out, 4);
}
